// MoELayer_10204842295660
// MI455X (gfx1250) — matrix operand order repaired and checked by reading, not compiled, not run on hardware
//
#include <hip/hip_runtime.h>
#include <hip/hip_bf16.h>

typedef __bf16 v16bf __attribute__((ext_vector_type(16)));
typedef __bf16 v8bf  __attribute__((ext_vector_type(8)));
typedef float  v8f   __attribute__((ext_vector_type(8)));
typedef unsigned int v4u __attribute__((ext_vector_type(4)));
typedef int v8i __attribute__((ext_vector_type(8)));
typedef int v4i __attribute__((ext_vector_type(4)));

#define B_   8
#define C_   64
#define E_   8
#define H_   128
#define W_   128
#define HW_  16384
#define KDIM 576                  // 9 taps * 64 channels, tap-major
#define XS_ROWS   4               // rows h0-1 .. h0+2 (2 output rows per block)
#define XS_ELEMS  (XS_ROWS*130*64)
#define AW_BYTES  (C_*KDIM*2)     // 73728

__device__ __forceinline__ unsigned short f2bf_bits(float f) {
    unsigned u = __builtin_bit_cast(unsigned, f);
    u += 0x7FFFu + ((u >> 16) & 1u);     // round-to-nearest-even
    return (unsigned short)(u >> 16);
}

// ---------------- Kernel 1: global average pool over H*W per (b,c) ------------
__global__ void gap_kernel(const float* __restrict__ x, float* __restrict__ gap) {
    __shared__ float red[256];
    const int bc = blockIdx.x;                    // 0..511  (b*64 + c)
    const float* p = x + (size_t)bc * HW_;
    float s = 0.f;
    for (int i = threadIdx.x; i < HW_; i += 256) s += p[i];
    red[threadIdx.x] = s;
    __syncthreads();
    for (int off = 128; off > 0; off >>= 1) {
        if (threadIdx.x < off) red[threadIdx.x] += red[threadIdx.x + off];
        __syncthreads();
    }
    if (threadIdx.x == 0) gap[bc] = red[0] * (1.0f / HW_);
}

// ---------------- Kernel 2: gate (linear + softmax + top2) + folded bias ------
__global__ void gate_kernel(const float* __restrict__ gap, const float* __restrict__ Wg,
                            const float* __restrict__ bg, const float* __restrict__ Kp,
                            const float* __restrict__ bexp,
                            float* __restrict__ wout, float* __restrict__ bcK) {
    __shared__ float lg[64];
    __shared__ float sw[64];
    const int tid = threadIdx.x;                  // 64 threads
    const int b = tid >> 3, e = tid & 7;
    float acc = bg[e];
    for (int c = 0; c < C_; ++c) acc += gap[b * C_ + c] * Wg[c * E_ + e];
    lg[tid] = acc;
    __syncthreads();
    if (e == 0) {                                 // one thread per batch item
        float m1 = -1e30f; int i1 = 0;
        for (int j = 0; j < E_; ++j) { float v = lg[b*E_+j]; if (v > m1) { m1 = v; i1 = j; } }
        float m2 = -1e30f; int i2 = 0;
        for (int j = 0; j < E_; ++j) { if (j == i1) continue; float v = lg[b*E_+j]; if (v > m2) { m2 = v; i2 = j; } }
        float den = 0.f;
        for (int j = 0; j < E_; ++j) den += __expf(lg[b*E_+j] - m1);
        for (int j = 0; j < E_; ++j) sw[b*E_+j] = 0.f;
        sw[b*E_+i1] = __expf(lg[b*E_+i1] - m1) / den;
        sw[b*E_+i2] = __expf(m2 - m1) / den;
    }
    __syncthreads();
    wout[tid] = sw[tid];
    const float Kv = Kp[0];
    for (int k = 0; k < 8; ++k) {                 // 512 folded-bias values
        const int idx = tid + k * 64;
        const int bb = idx >> 6, c = idx & 63;
        float s = 0.f;
        for (int ee = 0; ee < E_; ++ee) s += sw[bb*E_+ee] * bexp[ee * C_ + c];
        bcK[idx] = Kv * s;
    }
}

// --------- Kernel 3: fold experts -> per-batch bf16 filter A=[64 x 576] -------
// Output layout: WcBf[b][o][t*64 + ci], t = kh*3+kw  (tap-major K for implicit GEMM)
__global__ void combine_kernel(const float* __restrict__ w, const float* __restrict__ Kp,
                               const float* __restrict__ Wexp, unsigned short* __restrict__ WcBf) {
    const int idx = blockIdx.x * 256 + threadIdx.x;       // exactly 294912 threads
    const int b   = idx / (C_ * KDIM);
    const int rem = idx - b * (C_ * KDIM);
    const int o   = rem / KDIM;
    const int k   = rem - o * KDIM;
    const int t   = k >> 6;                                // tap 0..8
    const int ci  = k & 63;
    float s = 0.f;
#pragma unroll
    for (int e = 0; e < E_; ++e)
        s += w[b * E_ + e] * Wexp[(size_t)(((e * C_ + o) * C_ + ci) * 9 + t)];
    WcBf[idx] = f2bf_bits(Kp[0] * s);
}

// --------- Kernel 4: implicit-GEMM conv3x3 via v_wmma_f32_16x16x32_bf16 -------
// grid = (H/2=64, B=8), block = 256 (8 waves). Each block: 2 output rows.
// Wave = (r0 = wave>>2, wq = wave&3): output row h0+r0, cols [32*wq, 32*wq+32),
// all 64 c_out as 4 M-tiles -> 8 accumulators, A-frag reused across 2 N-tiles.
__global__ void moe_conv_kernel(const float* __restrict__ x,
                                const unsigned short* __restrict__ WcBf,
                                const float* __restrict__ bcK,
                                float* __restrict__ out) {
    extern __shared__ __align__(32) char smem[];
    __bf16* Aw = (__bf16*)smem;                           // 64*576 bf16   = 73728 B
    __bf16* Xs = (__bf16*)(smem + AW_BYTES);              // 4*130*64 bf16 = 66560 B

    const int h2  = blockIdx.x;                           // 0..63 -> rows 2*h2, 2*h2+1
    const int b   = blockIdx.y;
    const int tid = threadIdx.x;
    const int wave = tid >> 5, lane = tid & 31;

    // ---- stage folded weights (A matrix, row-major [M=64][K=576]) ----
    const unsigned short* wsrc = WcBf + (size_t)b * C_ * KDIM;
#if __has_builtin(__builtin_amdgcn_tensor_load_to_lds)
    if (wave == 0) {
        // Tensor Data Mover: 1-D tile of 18432 dwords, global -> LDS
        const unsigned long long ga = (unsigned long long)(const void*)wsrc;
        const unsigned lds_addr = (unsigned)(unsigned long long)(void*)Aw;
        v4u g0;
        g0[0] = 1u;                                        // count=1, is_restore=0, gather off
        g0[1] = lds_addr;                                  // lds_addr [63:32]
        g0[2] = (unsigned)ga;                              // global_addr low
        g0[3] = (unsigned)((ga >> 32) & 0x01FFFFFFull) | (2u << 30);  // addr hi | type=2
        v8i g1;
        g1[0] = (int)(2u << 16);                           // wg_mask=0, data_size=2 (4B)
        g1[1] = (int)(0x4800u << 16);                      // tensor_dim0[15:0]=18432
        g1[2] = (int)(1u << 16);                           // dim0 hi=0, tensor_dim1=1
        g1[3] = (int)(0x4800u << 16);                      // dim1 hi=0, tile_dim0=18432
        g1[4] = 0;                                         // tile_dim1=0, tile_dim2=0
        g1[5] = 18432;                                     // tensor_dim0_stride low
        g1[6] = 0;
        g1[7] = 0;
        v4i z4 = {};
        v8i z8 = {};
        __builtin_amdgcn_tensor_load_to_lds(g0, g1, z4, z4, z8, 0);
    }
#else
    {
        const uint4* ws4 = (const uint4*)wsrc;
        uint4* wdst = (uint4*)Aw;
        for (int i = tid; i < AW_BYTES / 16; i += 256) wdst[i] = ws4[i];
    }
#endif

    // ---- stage x halo slab as HWC bf16 with zero padding: [4][130][64] ----
    const size_t xb = (size_t)b * C_ * HW_;
    const int h0 = h2 * 2;
    for (int i = tid; i < XS_ELEMS; i += 256) {
        const int c   = i / (XS_ROWS * 130);
        const int rem = i - c * (XS_ROWS * 130);
        const int r   = rem / 130;                        // 0..3 -> rows h0-1 .. h0+2
        const int wp  = rem - r * 130;                    // 0..129 -> cols -1..128
        const int hh  = h0 - 1 + r;
        const int ww  = wp - 1;
        float v = 0.f;
        if (hh >= 0 && hh < H_ && ww >= 0 && ww < W_)
            v = x[xb + (size_t)c * HW_ + (size_t)hh * W_ + ww];
        Xs[(r * 130 + wp) * 64 + c] = __builtin_bit_cast(__bf16, f2bf_bits(v));
    }
#if __has_builtin(__builtin_amdgcn_tensor_load_to_lds)
    if (wave == 0) __builtin_amdgcn_s_wait_tensorcnt(0);
#endif
    __syncthreads();

    const int wq    = wave & 3;                           // column quarter
    const int r0    = wave >> 2;                          // which of 2 output rows
    const int wbase = wq * 32;
    const int n     = lane & 15;                          // column in 16-tile / A row
    const int g     = (lane < 16) ? 0 : 16;               // A-frag K-half select
    const int koff  = (lane < 16) ? 0 : 16;               // B-frag K-half select

    v8f acc[2][4] = {};                                   // [N-tile][M-tile]

#pragma unroll
    for (int t = 0; t < 9; ++t) {
        const int kh = t / 3, kw = t - kh * 3;
        const int prow = (r0 + kh) * 130;                 // input row in slab
#pragma unroll
        for (int ch = 0; ch < 2; ++ch) {
            const int c0    = ch * 32;
            const int kbase = t * 64 + c0;
            v16bf bf0 = *(const v16bf*)&Xs[(prow + wbase +      n + kw) * 64 + c0 + koff];
            v16bf bf1 = *(const v16bf*)&Xs[(prow + wbase + 16 + n + kw) * 64 + c0 + koff];
#pragma unroll
            for (int mt = 0; mt < 4; ++mt) {
                const __bf16* pa = &Aw[(mt * 16 + n) * KDIM + kbase + g];
                v8bf a0 = *(const v8bf*)pa;               // K = kbase+g .. +7
                v8bf a1 = *(const v8bf*)(pa + 8);         // K = kbase+g+8 .. +15
                v16bf afrag = __builtin_shufflevector(a0, a1,
                    0,1,2,3,4,5,6,7,8,9,10,11,12,13,14,15);
                acc[0][mt] = __builtin_amdgcn_wmma_f32_16x16x32_bf16(
                    false, afrag, false, bf0, (short)0, acc[0][mt], false, false);
                acc[1][mt] = __builtin_amdgcn_wmma_f32_16x16x32_bf16(
                    false, afrag, false, bf1, (short)0, acc[1][mt], false, false);
            }
        }
    }

    // ---- epilogue: out = x + conv (K folded) + K*bias ----
    const int hrow = h0 + r0;
    const int rofs = (lane < 16) ? 0 : 8;
#pragma unroll
    for (int nt = 0; nt < 2; ++nt) {
        const int wcol = wbase + nt * 16 + n;
#pragma unroll
        for (int mt = 0; mt < 4; ++mt) {
#pragma unroll
            for (int r = 0; r < 8; ++r) {
                const int c = mt * 16 + r + rofs;
                const size_t oi = xb + (size_t)c * HW_ + (size_t)hrow * W_ + wcol;
                out[oi] = x[oi] + acc[nt][mt][r] + bcK[b * C_ + c];
            }
        }
    }
}

extern "C" void kernel_launch(void* const* d_in, const int* in_sizes, int n_in,
                              void* d_out, int out_size, void* d_ws, size_t ws_size,
                              hipStream_t stream) {
    const float* x    = (const float*)d_in[0];
    const float* K    = (const float*)d_in[1];
    const float* Wg   = (const float*)d_in[2];
    const float* bg   = (const float*)d_in[3];
    const float* Wexp = (const float*)d_in[4];
    const float* bexp = (const float*)d_in[5];
    float* out = (float*)d_out;

    // workspace layout
    float* gap = (float*)d_ws;                               // 512 f32
    float* w   = gap + 512;                                  // 64  f32
    float* bcK = w + 64;                                     // 512 f32
    unsigned short* WcBf = (unsigned short*)((char*)d_ws + 4352);  // 8*64*576 bf16

    gap_kernel<<<B_ * C_, 256, 0, stream>>>(x, gap);
    gate_kernel<<<1, 64, 0, stream>>>(gap, Wg, bg, K, bexp, w, bcK);
    combine_kernel<<<(B_ * C_ * KDIM) / 256, 256, 0, stream>>>(w, K, Wexp, WcBf);

    const size_t smem = (size_t)AW_BYTES + (size_t)(XS_ELEMS * 2);  // 140288 B
    dim3 grid(H_ / 2, B_);
    moe_conv_kernel<<<grid, 256, smem, stream>>>(x, WcBf, bcK, out);
}
